// GLCM_58591943852361
// MI455X (gfx1250) — matrix operand order repaired and checked by reading, not compiled, not run on hardware
//
#include <hip/hip_runtime.h>

typedef __attribute__((ext_vector_type(16))) _Float16 v16h;
typedef __attribute__((ext_vector_type(8)))  _Float16 v8h;
typedef __attribute__((ext_vector_type(8)))  float    v8f;

#define NDIR 8
#define GL   32   // gray levels

// Shared memory carve (bytes):
//   [0,     18432)  Pf: 9 x 1024 f16 (rows 0..7 = GLCMs, row 8 = zeros);
//                   first 16384 B doubles as the 64x64 int32 image in phase 0/1
//   [18432, 51200)  cnt: 8 x 1024 u32 co-occurrence counts;
//                   first 16384 B reused as PfT (1024 x 8 f16) after phase 2a
//   [51200, 51328)  hcnt: 32 u32 value histogram
//   [51328, 52352)  Dw: 16x16 f32 (A x W result)
//   [52352, 53376)  DA: 16x16 f32 (A x A^T result)
//   [53376, 53380)  sTot: hist normalizer
#define SMEM_BYTES 53384

__global__ __launch_bounds__(256)
void glcm_feat_kernel(const int* __restrict__ q, float* __restrict__ out) {
  __shared__ __align__(16) unsigned char smem[SMEM_BYTES];
  int*      img  = (int*)smem;
  _Float16* Pf   = (_Float16*)smem;              // 9 rows, row 8 zero pad
  unsigned* cnt  = (unsigned*)(smem + 18432);
  _Float16* PfT  = (_Float16*)(smem + 18432);    // overlays cnt after phase 2a
  unsigned* hcnt = (unsigned*)(smem + 51200);
  float*    Dw   = (float*)(smem + 51328);
  float*    DA   = (float*)(smem + 52352);
  float*    sTot = (float*)(smem + 53376);

  const int tid = threadIdx.x;
  const int n   = blockIdx.x;          // image index, 0..4095

  // ---- phase 0: zero counters + Pf pad row, load image into LDS ----
  for (int x = tid; x < 8192; x += 256) cnt[x] = 0u;
  if (tid < 32) hcnt[tid] = 0u;
  {
    unsigned* pad = (unsigned*)(smem + 16384);   // Pf row 8 (2048 B)
    for (int x = tid; x < 512; x += 256) pad[x] = 0u;
    const int4* qv = (const int4*)(q + (size_t)n * 4096);
    int4* iv = (int4*)img;
    for (int x = tid; x < 1024; x += 256) iv[x] = qv[x];
  }
  __syncthreads();

  // ---- phase 1: value histogram + co-occurrence counting (LDS atomics) ----
  for (int x = tid; x < 4096; x += 256) atomicAdd(&hcnt[img[x] & 31], 1u);

  // (dr,dc) per reference: Python round() collapses d=2 diagonals onto d=1
  constexpr int DR[NDIR] = {0, 1, 1,  1, 0, 1, 2,  1};
  constexpr int DC[NDIR] = {1, 1, 0, -1, 2, 1, 0, -1};
#pragma unroll
  for (int d = 0; d < NDIR; ++d) {
    const int dr = DR[d], dc = DC[d];
    const int c0 = (dc < 0) ? -dc : 0;           // dr >= 0 for all dirs
    const int ch = 64 - ((dc > 0) ? dc : 0);
    const int Np = (64 - dr) << 6;               // rows * padded width 64
    for (int p = tid; p < Np; p += 256) {
      const int r = p >> 6, c = p & 63;          // no div/mod
      if (c >= c0 && c < ch) {
        const int i = img[r * 64 + c];
        const int j = img[(r + dr) * 64 + (c + dc)];
        atomicAdd(&cnt[(d << 10) + i * GL + j], 1u);
      }
    }
  }
  __syncthreads();

  // ---- phase 2a: hist total; symmetrize+normalize counts -> f16 (over img) ----
  if (tid == 0) {
    unsigned t = 0;
    for (int v = 1; v < 32; ++v) t += hcnt[v];
    *sTot = (float)t;
  }
  // 1 / (2 * num_pairs) per direction
  constexpr float invTot[NDIR] = {
      1.0f / 8064.0f, 1.0f / 7938.0f, 1.0f / 8064.0f, 1.0f / 7938.0f,
      1.0f / 7936.0f, 1.0f / 7938.0f, 1.0f / 7936.0f, 1.0f / 7938.0f};
#pragma unroll
  for (int d = 0; d < NDIR; ++d) {
    const float inv = invTot[d];
    for (int ij = tid; ij < 1024; ij += 256) {
      const int i = ij >> 5, j = ij & 31;
      const float pv =
          (float)(cnt[(d << 10) + ij] + cnt[(d << 10) + j * GL + i]) * inv;
      Pf[(d << 10) + ij] = (_Float16)pv;
    }
  }
  __syncthreads();

  // ---- phase 2b: k-major transpose PfT[k][d] = Pf[d][k] (overwrites cnt) ----
  for (int x = tid; x < 8192; x += 256) {
    const int k = x >> 3, d = x & 7;
    PfT[x] = Pf[(d << 10) + k];
  }
  __syncthreads();

  // ---- phase 3: wave 0 computes both K=1024 GEMMs with v_wmma_f32_16x16x32_f16
  //   A  (16x1024): rows 0..7 = P per direction, rows 8..15 = 0 (pad row)
  //   Dw = A x W   (W cols: w2, |d|, 1/(1+w2), i, j, i^2, j^2, i*j, 0...)
  //   DA = A x A^T (diagonal = ASM per direction)
  if (tid < 32) {
    const int ln   = tid;
    const int M    = ln & 15;                 // A row for this lane
    const int row  = (M < 8) ? M : 8;         // clamp to zero pad row
    const int koff = (ln & 16) ? 8 : 0;       // K sub-offset per ISA A layout
    const _Float16* ap0 = Pf + (row << 10) + koff;
    v8f accW = {};
    v8f accA = {};
    for (int t = 0; t < 32; ++t) {
      const int kk0 = t << 5;
      // A fragment: halves 0..7 -> K=kk0+koff+0..7, halves 8..15 -> +16
      const _Float16* ap = ap0 + kk0;
      v8h alo = *(const v8h*)(ap);
      v8h ahi = *(const v8h*)(ap + 16);
      v16h a = __builtin_shufflevector(alo, ahi, 0, 1, 2, 3, 4, 5, 6, 7,
                                       8, 9, 10, 11, 12, 13, 14, 15);
      // B fragment for weights: lane = N column, halves = K in same order as A
      v16h b = {};
      for (int s = 0; s < 16; ++s) {
        const int K  = kk0 + koff + ((s < 8) ? s : s + 8);
        const float fi = (float)(K >> 5);
        const float fj = (float)(K & 31);
        const float fd = fi - fj;
        const float w2 = fd * fd;
        float v = 0.0f;
        switch (M) {
          case 0: v = w2; break;
          case 1: v = __builtin_fabsf(fd); break;
          case 2: v = __builtin_amdgcn_rcpf(1.0f + w2); break;
          case 3: v = fi; break;
          case 4: v = fj; break;
          case 5: v = fi * fi; break;
          case 6: v = fj * fj; break;
          case 7: v = fi * fj; break;
          default: break;
        }
        b[s] = (_Float16)v;
      }
      // B fragment for A^T: column n = row n of Pf, same K order as A
      v16h b2 = a;

      accW = __builtin_amdgcn_wmma_f32_16x16x32_f16(false, a, false, b,
                                                    (short)0, accW, false, false);
      accA = __builtin_amdgcn_wmma_f32_16x16x32_f16(false, a, false, b2,
                                                    (short)0, accA, false, false);
    }
    // D layout: lanes 0..15 hold N=lane, VGPR r -> M=r (rows 0..7 are ours)
    if (ln < 16) {
#pragma unroll
      for (int r = 0; r < 8; ++r) {
        Dw[r * 16 + ln] = accW[r];
        DA[r * 16 + ln] = accA[r];
      }
    }
  }
  __syncthreads();

  // ---- phase 4: derive features and scatter 79 outputs ----
  // output index: out[b][f][h][w], n = ((b*16)+h)*16 + w
  const int w  = n & 15;
  const int hh = (n >> 4) & 15;
  const int bb = n >> 8;
  float* ob = out + (size_t)bb * 79 * 256 + hh * 16 + w;

  if (tid < 8) {
    const int k = tid;
    const float contrast = Dw[k * 16 + 0];
    const float dissim   = Dw[k * 16 + 1];
    const float homog    = Dw[k * 16 + 2];
    const float mui      = Dw[k * 16 + 3];
    const float muj      = Dw[k * 16 + 4];
    const float sii      = Dw[k * 16 + 5];
    const float sjj      = Dw[k * 16 + 6];
    const float sij      = Dw[k * 16 + 7];
    const float asmv     = DA[k * 16 + k];
    float vi = sii - mui * mui; vi = vi > 0.0f ? vi : 0.0f;
    float vj = sjj - muj * muj; vj = vj > 0.0f ? vj : 0.0f;
    const float si = __builtin_sqrtf(vi);
    const float sj = __builtin_sqrtf(vj);
    const float cov = sij - mui * muj;
    const float tiny = 1e-15f;
    const float corr = (si < tiny || sj < tiny) ? 1.0f : cov / (si * sj);
    const float energy = __builtin_sqrtf(asmv);
    ob[(31 + k) * 256] = contrast;
    ob[(39 + k) * 256] = dissim;
    ob[(47 + k) * 256] = homog;
    ob[(55 + k) * 256] = energy;
    ob[(63 + k) * 256] = corr;
    ob[(71 + k) * 256] = asmv;
  } else if (tid >= 64 && tid < 95) {
    const int v = tid - 63;               // levels 1..31
    const float tot = *sTot;
    const float hv = tot > 0.0f ? (float)hcnt[v] / tot : 0.0f;
    ob[(v - 1) * 256] = hv;
  }
}

extern "C" void kernel_launch(void* const* d_in, const int* in_sizes, int n_in,
                              void* d_out, int out_size, void* d_ws, size_t ws_size,
                              hipStream_t stream) {
  (void)n_in; (void)out_size; (void)d_ws; (void)ws_size;
  const int* q = (const int*)d_in[0];
  float* out = (float*)d_out;
  const int n_imgs = in_sizes[0] / 4096;   // 4096 images of 64x64
  glcm_feat_kernel<<<dim3(n_imgs), dim3(256), 0, stream>>>(q, out);
}
